// DeepSeekMoEV3_64278480552168
// MI455X (gfx1250) — compile-verified, matrix operand order checked by reading, not run on hardware
//
#include <hip/hip_runtime.h>
#include <math.h>

typedef __attribute__((ext_vector_type(16))) __bf16 v16bf;
typedef __attribute__((ext_vector_type(8)))  __bf16 v8bf;
typedef __attribute__((ext_vector_type(4)))  __bf16 v4bf;
typedef __attribute__((ext_vector_type(8)))  float  v8f;

#define LDA 40  // elements; 80B rows keep every 16B chunk aligned, conflict-free
#define LDB 40  // column-major B tile: Bs[n*LDB + k]

// ---------------------------------------------------------------------------
// Utility kernels (vectorized: 16B global accesses)
// ---------------------------------------------------------------------------
__global__ void f32_to_bf16_kernel(const float* __restrict__ in,
                                   __bf16* __restrict__ out, long n4) {
    // n4 = number of float4 chunks
    long i = (long)blockIdx.x * blockDim.x + threadIdx.x;
    long stride = (long)gridDim.x * blockDim.x;
    for (; i < n4; i += stride) {
        float4 f = *(const float4*)(in + i * 4);
        v4bf o = {(__bf16)f.x, (__bf16)f.y, (__bf16)f.z, (__bf16)f.w};
        *(v4bf*)(out + i * 4) = o;
    }
}

// Normalize G group centroids followed by E expert centroids into `out` (40 x D)
__global__ void norm_rows_kernel(const float* __restrict__ gc,
                                 const float* __restrict__ ec,
                                 float* __restrict__ out, int G, int D) {
    int b = blockIdx.x;
    const float* src = (b < G) ? gc + (long)b * D : ec + (long)(b - G) * D;
    __shared__ float part[8];
    __shared__ float s_inv;
    int tid = threadIdx.x, wid = tid >> 5, lane = tid & 31;
    float p = 0.f;
    for (int i = tid; i < D; i += 256) { float v = src[i]; p += v * v; }
    for (int off = 16; off; off >>= 1) p += __shfl_xor(p, off, 32);
    if (lane == 0) part[wid] = p;
    __syncthreads();
    if (tid == 0) {
        float ss = 0.f;
        for (int w = 0; w < 8; w++) ss += part[w];
        s_inv = 1.0f / fmaxf(sqrtf(ss), 1e-12f);
    }
    __syncthreads();
    float inv = s_inv;
    for (int i = tid; i < D; i += 256) out[(long)b * D + i] = src[i] * inv;
}

// ---------------------------------------------------------------------------
// Routing: group scores -> top-4 groups -> masked expert scores -> top-4 -> softmax
// ---------------------------------------------------------------------------
__global__ void routing_kernel(const float* __restrict__ x,
                               const float* __restrict__ cent, // (G+E) x D, normalized
                               const float* __restrict__ bias,
                               int* __restrict__ top_idx,      // N*K
                               float* __restrict__ gates) {    // N*K
    const int D = 1024, G = 8, E = 32, K = 4;
    __shared__ float xsh[1024];
    __shared__ float sc[41]; // 0..39 dots, 40 = sum of squares
    int t = blockIdx.x, tid = threadIdx.x;
    int wid = tid >> 5, lane = tid & 31;
    const float* xr = x + (long)t * D;
    for (int i = tid; i < D; i += 256) xsh[i] = xr[i];
    __syncthreads();
    for (int task = wid; task < 41; task += 8) {
        float p = 0.f;
        if (task < 40) {
            const float* c = cent + (long)task * D;
            for (int i = lane; i < D; i += 32) p += xsh[i] * c[i];
        } else {
            for (int i = lane; i < D; i += 32) p += xsh[i] * xsh[i];
        }
        for (int off = 16; off; off >>= 1) p += __shfl_xor(p, off, 32);
        if (lane == 0) sc[task] = p;
    }
    __syncthreads();
    if (tid == 0) {
        float inv = 1.0f / fmaxf(sqrtf(sc[40]), 1e-12f);
        bool gkeep[G];
        for (int g = 0; g < G; g++) gkeep[g] = false;
        for (int it = 0; it < 4; it++) {
            int best = -1; float bv = -3.4e38f;
            for (int g = 0; g < G; g++)
                if (!gkeep[g] && sc[g] * inv > bv) { bv = sc[g] * inv; best = g; }
            gkeep[best] = true;
        }
        float es[E];
        for (int e = 0; e < E; e++) {
            float v = sc[G + e] * inv + bias[e];
            es[e] = gkeep[e >> 2] ? v : -1e9f; // E/G = 4 experts per group
        }
        int   idx[K];
        float val[K];
        bool used[E];
        for (int e = 0; e < E; e++) used[e] = false;
        for (int k = 0; k < K; k++) {
            int best = -1; float bv = -3.4e38f;
            for (int e = 0; e < E; e++)
                if (!used[e] && es[e] > bv) { bv = es[e]; best = e; }
            used[best] = true; idx[k] = best; val[k] = bv;
        }
        float m = val[0];
        float se = 0.f;
        float ex[K];
        for (int k = 0; k < K; k++) { ex[k] = __expf(val[k] - m); se += ex[k]; }
        float rs = 1.0f / se;
        for (int k = 0; k < K; k++) {
            top_idx[t * K + k] = idx[k];
            gates[t * K + k] = ex[k] * rs;
        }
    }
}

// ---------------------------------------------------------------------------
// Capacity assignment: one wave per expert scans all slots in flat order.
// ---------------------------------------------------------------------------
__global__ void capacity_kernel(const int* __restrict__ slot_e,
                                int* __restrict__ slot_pos, int nslots, int cap) {
    int e = blockIdx.x;
    int lane = threadIdx.x; // 32 threads
    unsigned lmask = (1u << lane) - 1u;
    int base = 0;
    for (int i0 = 0; i0 < nslots; i0 += 32) {
        int i = i0 + lane;
        bool m = (slot_e[i] == e);
        unsigned msk = (unsigned)__ballot(m);
        if (m) {
            int p = base + __popc(msk & lmask);
            slot_pos[i] = (p < cap) ? p : -1;
        }
        base += __popc(msk);
    }
}

// ---------------------------------------------------------------------------
// Gather kept tokens into per-expert capacity buffer (bf16). 256 thr x 4 elems.
// ---------------------------------------------------------------------------
__global__ void build_buf_kernel(const float* __restrict__ x,
                                 const int* __restrict__ slot_e,
                                 const int* __restrict__ slot_pos,
                                 __bf16* __restrict__ buf, int cap, int D) {
    int s = blockIdx.x;
    int p = slot_pos[s];
    if (p < 0) return;
    int e = slot_e[s];
    int tok = s >> 2; // K = 4
    const float* xr = x + (long)tok * D;
    __bf16* dst = buf + ((long)e * cap + p) * D;
    int c = threadIdx.x << 2; // D = 1024 = 256*4 exactly
    float4 f = *(const float4*)(xr + c);
    v4bf o = {(__bf16)f.x, (__bf16)f.y, (__bf16)f.z, (__bf16)f.w};
    *(v4bf*)(dst + c) = o;
}

// ---------------------------------------------------------------------------
// Fragment load: two aligned 16B LDS chunks -> v16bf (compiles to ds_load_b128).
// K(j) = (j<8?0:16) + half*8 + (j&7)  (ISA §7.12.2, 16-bit A/B layouts)
// ---------------------------------------------------------------------------
__device__ __forceinline__ v16bf load_frag(const __bf16* base, int row_off, int hl) {
    v8bf lo = *(const v8bf*)(base + row_off + (hl << 3));
    v8bf hi = *(const v8bf*)(base + row_off + 16 + (hl << 3));
    return __builtin_shufflevector(lo, hi, 0, 1, 2, 3, 4, 5, 6, 7,
                                   8, 9, 10, 11, 12, 13, 14, 15);
}

// ---------------------------------------------------------------------------
// WMMA GEMM: C[M,N] = A[M,K] (bf16) x B[K,N] (fp32 -> bf16, transposed in LDS)
// FUSED: dual-B, C = (silu(A*B1) * (A*B2)) as bf16.  !FUSED: C = A*B1 as fp32.
// Block: 256 threads = 8 waves; block tile 64x64; wave tile 32x16; K-step 32.
// Software-pipelined: global loads for step k+1 issued before WMMAs of step k,
// so the load-wait overlaps matrix math instead of stalling the LDS store.
// ---------------------------------------------------------------------------
template <bool FUSED, typename OutT>
__global__ __launch_bounds__(256) void gemm_wmma_kernel(
    const __bf16* __restrict__ Ag, const float* __restrict__ B1g,
    const float* __restrict__ B2g, OutT* __restrict__ Cg,
    int M, int N, int Kd, long sA, long sB, long sC) {
    const __bf16* A = Ag + (long)blockIdx.z * sA;
    const float* B1 = B1g + (long)blockIdx.z * sB;
    const float* B2 = FUSED ? (B2g + (long)blockIdx.z * sB) : nullptr;
    OutT* C = Cg + (long)blockIdx.z * sC;

    __shared__ alignas(16) __bf16 As[64 * LDA];                 // 64 M-rows x 32 K
    __shared__ alignas(16) __bf16 Bs1[64 * LDB];                // 64 N-cols x 32 K
    __shared__ alignas(16) __bf16 Bs2[FUSED ? 64 * LDB : 8];

    int tid = threadIdx.x;
    int wid = tid >> 5, lane = tid & 31;
    int hl = lane >> 4, r = lane & 15;
    int wm = (wid >> 2) << 5; // 0 or 32   (wave covers 32 M-rows)
    int wn = (wid & 3) << 4;  // 0,16,32,48 (wave covers 16 N-cols)
    int mBase = blockIdx.y * 64;
    int nBase = blockIdx.x * 64;

    // Per-thread staging assignments (fixed across K-steps)
    int aRow = tid >> 2, aCol = (tid & 3) << 3;      // A: one 16B chunk
    int bRow0 = tid >> 4, bCol0 = (tid & 15) << 2;   // B chunk 0 (tid)
    int bRow1 = (tid + 256) >> 4, bCol1 = bCol0;     // B chunk 1 (tid+256)

    v8bf  aReg;
    float4 b1Reg0, b1Reg1, b2Reg0, b2Reg1;

    auto gload = [&](int kk) {
        aReg = *(const v8bf*)(A + (long)(mBase + aRow) * Kd + kk + aCol);
        long gb0 = (long)(kk + bRow0) * N + nBase + bCol0;
        long gb1 = (long)(kk + bRow1) * N + nBase + bCol1;
        b1Reg0 = *(const float4*)(B1 + gb0);
        b1Reg1 = *(const float4*)(B1 + gb1);
        if constexpr (FUSED) {
            b2Reg0 = *(const float4*)(B2 + gb0);
            b2Reg1 = *(const float4*)(B2 + gb1);
        }
        if (kk + 64 < Kd) { // keep next-next tile flowing into L2
            __builtin_prefetch(B1 + gb0 + 64 * (long)N, 0, 0);
            if constexpr (FUSED) __builtin_prefetch(B2 + gb0 + 64 * (long)N, 0, 0);
        }
    };
    auto lstore = [&]() {
        *(v8bf*)(As + aRow * LDA + aCol) = aReg;
        Bs1[(bCol0 + 0) * LDB + bRow0] = (__bf16)b1Reg0.x;
        Bs1[(bCol0 + 1) * LDB + bRow0] = (__bf16)b1Reg0.y;
        Bs1[(bCol0 + 2) * LDB + bRow0] = (__bf16)b1Reg0.z;
        Bs1[(bCol0 + 3) * LDB + bRow0] = (__bf16)b1Reg0.w;
        Bs1[(bCol1 + 0) * LDB + bRow1] = (__bf16)b1Reg1.x;
        Bs1[(bCol1 + 1) * LDB + bRow1] = (__bf16)b1Reg1.y;
        Bs1[(bCol1 + 2) * LDB + bRow1] = (__bf16)b1Reg1.z;
        Bs1[(bCol1 + 3) * LDB + bRow1] = (__bf16)b1Reg1.w;
        if constexpr (FUSED) {
            Bs2[(bCol0 + 0) * LDB + bRow0] = (__bf16)b2Reg0.x;
            Bs2[(bCol0 + 1) * LDB + bRow0] = (__bf16)b2Reg0.y;
            Bs2[(bCol0 + 2) * LDB + bRow0] = (__bf16)b2Reg0.z;
            Bs2[(bCol0 + 3) * LDB + bRow0] = (__bf16)b2Reg0.w;
            Bs2[(bCol1 + 0) * LDB + bRow1] = (__bf16)b2Reg1.x;
            Bs2[(bCol1 + 1) * LDB + bRow1] = (__bf16)b2Reg1.y;
            Bs2[(bCol1 + 2) * LDB + bRow1] = (__bf16)b2Reg1.z;
            Bs2[(bCol1 + 3) * LDB + bRow1] = (__bf16)b2Reg1.w;
        }
    };

    v8f acc1[2] = {{}, {}};
    v8f acc2[2] = {{}, {}};

    gload(0); // prologue
    for (int kk = 0; kk < Kd; kk += 32) {
        lstore();
        __syncthreads();
        if (kk + 32 < Kd) gload(kk + 32); // overlap with WMMAs below

        v16bf b1 = load_frag(Bs1, (wn + r) * LDB, hl);
        v16bf b2;
        if constexpr (FUSED) b2 = load_frag(Bs2, (wn + r) * LDB, hl);
#pragma unroll
        for (int t = 0; t < 2; t++) {
            v16bf a = load_frag(As, (wm + t * 16 + r) * LDA, hl);
            acc1[t] = __builtin_amdgcn_wmma_f32_16x16x32_bf16(
                false, a, false, b1, (short)0, acc1[t], false, false);
            if constexpr (FUSED)
                acc2[t] = __builtin_amdgcn_wmma_f32_16x16x32_bf16(
                    false, a, false, b2, (short)0, acc2[t], false, false);
        }
        __syncthreads();
    }

#pragma unroll
    for (int t = 0; t < 2; t++) {
#pragma unroll
        for (int i = 0; i < 8; i++) {
            int gm = mBase + wm + t * 16 + (hl << 3) + i;
            int gn = nBase + wn + r;
            long o = (long)gm * N + gn;
            if constexpr (FUSED) {
                float g = acc1[t][i];
                float s = g / (1.0f + __expf(-g)); // silu
                C[o] = (OutT)(s * acc2[t][i]);
            } else {
                C[o] = (OutT)acc1[t][i];
            }
        }
    }
}

// ---------------------------------------------------------------------------
// Scatter: per token, accumulate its 4 gated expert outputs onto d_out.
// Vectorized: one float4 column chunk per thread (D = 1024 = 256*4).
// ---------------------------------------------------------------------------
__global__ void scatter_add_kernel(const int* __restrict__ slot_e,
                                   const int* __restrict__ slot_pos,
                                   const float* __restrict__ gates,
                                   const float* __restrict__ out_buf,
                                   float* __restrict__ out, int cap, int D) {
    int t = blockIdx.x;
    int c = threadIdx.x << 2;
    float4 sum = {0.f, 0.f, 0.f, 0.f};
    for (int k = 0; k < 4; k++) {
        int s = t * 4 + k;
        int p = slot_pos[s];
        if (p >= 0) {
            int e = slot_e[s];
            float g = gates[s];
            float4 v = *(const float4*)(out_buf + ((long)e * cap + p) * D + c);
            sum.x += g * v.x; sum.y += g * v.y;
            sum.z += g * v.z; sum.w += g * v.w;
        }
    }
    float4 o = *(const float4*)(out + (long)t * D + c);
    o.x += sum.x; o.y += sum.y; o.z += sum.z; o.w += sum.w;
    *(float4*)(out + (long)t * D + c) = o;
}

// ---------------------------------------------------------------------------
// Host launcher
// ---------------------------------------------------------------------------
extern "C" void kernel_launch(void* const* d_in, const int* in_sizes, int n_in,
                              void* d_out, int out_size, void* d_ws, size_t ws_size,
                              hipStream_t stream) {
    const float* x    = (const float*)d_in[0];
    const float* gc   = (const float*)d_in[1];
    const float* ec   = (const float*)d_in[2];
    const float* bias = (const float*)d_in[3];
    const float* wg   = (const float*)d_in[4];
    const float* wu   = (const float*)d_in[5];
    const float* wd   = (const float*)d_in[6];
    const float* sg   = (const float*)d_in[7];
    const float* su   = (const float*)d_in[8];
    const float* sd   = (const float*)d_in[9];

    const int N = 4096, D = 1024, E = 32, G = 8, K = 4, H = 2048, SH = 2048;
    const int CAP = 640; // int(N/E * K * 1.25)
    const int NK = N * K;

    char* ws = (char*)d_ws;
    size_t off = 0;
    auto wsalloc = [&](size_t bytes) -> void* {
        void* p = ws + off;
        off += (bytes + 255) & ~(size_t)255;
        return p;
    };
    float*  cent     = (float*)wsalloc((size_t)(G + E) * D * 4);
    int*    slot_e   = (int*)wsalloc((size_t)NK * 4);
    float*  gates    = (float*)wsalloc((size_t)NK * 4);
    int*    slot_pos = (int*)wsalloc((size_t)NK * 4);
    __bf16* x16      = (__bf16*)wsalloc((size_t)N * D * 2);
    __bf16* hs       = (__bf16*)wsalloc((size_t)N * SH * 2);
    __bf16* buf      = (__bf16*)wsalloc((size_t)E * CAP * D * 2);
    __bf16* hexp     = (__bf16*)wsalloc((size_t)E * CAP * H * 2);
    float*  outb     = (float*)wsalloc((size_t)E * CAP * D * 4);

    // Routing pipeline
    norm_rows_kernel<<<G + E, 256, 0, stream>>>(gc, ec, cent, G, D);
    f32_to_bf16_kernel<<<1024, 256, 0, stream>>>(x, x16, (long)N * D / 4);
    routing_kernel<<<N, 256, 0, stream>>>(x, cent, bias, slot_e, gates);
    capacity_kernel<<<E, 32, 0, stream>>>(slot_e, slot_pos, NK, CAP);
    build_buf_kernel<<<NK, 256, 0, stream>>>(x, slot_e, slot_pos, buf, CAP, D);

    // Shared expert: hs = silu(x@sg)*(x@su); d_out = hs@sd
    gemm_wmma_kernel<true, __bf16><<<dim3(SH / 64, N / 64, 1), 256, 0, stream>>>(
        x16, sg, su, hs, N, SH, D, 0, 0, 0);
    gemm_wmma_kernel<false, float><<<dim3(D / 64, N / 64, 1), 256, 0, stream>>>(
        hs, sd, nullptr, (float*)d_out, N, D, SH, 0, 0, 0);

    // Routed experts: hexp = silu(buf@wg)*(buf@wu); outb = hexp@wd
    gemm_wmma_kernel<true, __bf16><<<dim3(H / 64, CAP / 64, E), 256, 0, stream>>>(
        buf, wg, wu, hexp, CAP, H, D, (long)CAP * D, (long)D * H, (long)CAP * H);
    gemm_wmma_kernel<false, float><<<dim3(D / 64, CAP / 64, E), 256, 0, stream>>>(
        hexp, wd, nullptr, outb, CAP, D, H, (long)CAP * H, (long)H * D, (long)CAP * D);

    // Combine
    scatter_add_kernel<<<N, 256, 0, stream>>>(slot_e, slot_pos, gates, outb,
                                              (float*)d_out, CAP, D);
}
